// ResearchTGN_8830452760715
// MI455X (gfx1250) — matrix operand order repaired and checked by reading, not compiled, not run on hardware
//
#include <hip/hip_runtime.h>
#include <hip/hip_bf16.h>
#include <stdint.h>

// ---------------------------------------------------------------------------
// Types for WMMA / TDM
// ---------------------------------------------------------------------------
typedef __attribute__((ext_vector_type(16))) __bf16 v16bf;
typedef __attribute__((ext_vector_type(8)))  float  v8f;
typedef __attribute__((ext_vector_type(4)))  unsigned int u32x4;
typedef __attribute__((ext_vector_type(8)))  int          i32x8;
typedef __attribute__((ext_vector_type(4)))  int          i32x4;

union FragAB {
    v16bf v;
    uint4 u[2];
};

#define DEV static __device__ __forceinline__

#if defined(__has_builtin)
#if __has_builtin(__builtin_amdgcn_tensor_load_to_lds) && \
    __has_builtin(__builtin_amdgcn_s_wait_tensorcnt)
#define USE_TDM 1
#endif
#endif
#ifndef USE_TDM
#define USE_TDM 0
#endif

// ---------------------------------------------------------------------------
// bf16 helpers (explicit RNE conversion)
// ---------------------------------------------------------------------------
DEV uint16_t f2bf_bits(float f) {
    uint32_t u = __builtin_bit_cast(uint32_t, f);
    uint32_t r = (u + 0x7fffu + ((u >> 16) & 1u)) >> 16;
    return (uint16_t)r;
}
DEV __bf16 f2bf(float f) {
    uint16_t b = f2bf_bits(f);
    return __builtin_bit_cast(__bf16, b);
}
DEV float bf_lo(uint32_t packed) {
    uint32_t u = packed << 16;
    return __builtin_bit_cast(float, u);
}
DEV float bf_hi(uint32_t packed) {
    uint32_t u = packed & 0xffff0000u;
    return __builtin_bit_cast(float, u);
}
DEV uint32_t pack2bf(float a, float b) {
    return (uint32_t)f2bf_bits(a) | ((uint32_t)f2bf_bits(b) << 16);
}

// order-preserving int encoding of float for atomicMax-based segment max
DEV unsigned enc_f32(float f) {
    int i = __builtin_bit_cast(int, f);
    return (i >= 0) ? ((unsigned)i | 0x80000000u) : ~(unsigned)i;
}
DEV float dec_f32(unsigned k) {
    unsigned b = (k & 0x80000000u) ? (k & 0x7fffffffu) : ~k;
    return __builtin_bit_cast(float, b);
}

// ---------------------------------------------------------------------------
// Constants
// ---------------------------------------------------------------------------
#define EMB      128
#define HEADS    4
#define EDGE_D   130
#define EDGE_KP  160
#define SCALE_F  0.17677669529663689f   // 1/sqrt(32)

// ---------------------------------------------------------------------------
// Weight transpose + bf16 convert:  W[K,128] f32 -> Wt[128][KP] bf16 (zero pad)
// ---------------------------------------------------------------------------
__global__ void wconv_kernel(const float* __restrict__ W, __bf16* __restrict__ out,
                             int K, int KP) {
    int tid = blockIdx.x * blockDim.x + threadIdx.x;
    if (tid >= 128 * KP) return;
    int col = tid / KP;
    int kk  = tid - col * KP;
    float v = (kk < K) ? W[(size_t)kk * 128 + col] : 0.0f;
    out[tid] = f2bf(v);
}

// edge_attr [E,130] f32 -> [E,160] bf16 zero-padded
__global__ void padcvt_kernel(const float* __restrict__ src, __bf16* __restrict__ dst,
                              int rows, int K, int KP) {
    size_t tid = (size_t)blockIdx.x * blockDim.x + threadIdx.x;
    size_t total = (size_t)rows * KP;
    if (tid >= total) return;
    size_t r = tid / KP;
    int    c = (int)(tid - r * KP);
    float v = (c < K) ? src[r * K + c] : 0.0f;
    dst[tid] = f2bf(v);
}

// x = memory_table[n_id] gather, converted straight to bf16 [N,128]
__global__ void gathercvt_kernel(const float* __restrict__ table,
                                 const int* __restrict__ n_id,
                                 __bf16* __restrict__ xb, int N) {
    int tid = blockIdx.x * blockDim.x + threadIdx.x;
    if (tid >= N * 32) return;
    int i   = tid >> 5;
    int grp = tid & 31;
    const float4 val = *(const float4*)(table + (size_t)n_id[i] * EMB + grp * 4);
    uint2 packed;
    packed.x = pack2bf(val.x, val.y);
    packed.y = pack2bf(val.z, val.w);
    *(uint2*)(xb + (size_t)i * EMB + grp * 4) = packed;
}

// h (f32) -> relu -> bf16 (layer-2 input)
__global__ void relucvt_kernel(const float* __restrict__ h, __bf16* __restrict__ xb,
                               size_t n) {
    size_t tid = (size_t)blockIdx.x * blockDim.x + threadIdx.x;
    if (tid >= n) return;
    float v = h[tid];
    xb[tid] = f2bf(v > 0.0f ? v : 0.0f);
}

// ---------------------------------------------------------------------------
// bf16 WMMA GEMM:  D[M,128] = A[M,KP] * Bt^T + bias
//   A  : row-major bf16, KP multiple of 32
//   Bt : [128 cols][KP] bf16 (K contiguous per output column)
//   D  : f32 (OUT_BF16=0) or bf16 (OUT_BF16=1), [M,128]
// Block = 256 threads (8 waves); one 16-row tile per block; wave w owns output
// columns [16w,16w+16). A tile staged into LDS by the Tensor Data Mover
// (tensor_load_to_lds, issued by wave 0 only; OOB tail rows zero-fill), then
// fragments are two ds_load_b128 (A) / two global_load_b128 (B) per K-step.
// ---------------------------------------------------------------------------
template <int KP, int OUT_BF16>
__global__ __launch_bounds__(256)
void gemm_bf16_kernel(const __bf16* __restrict__ A,
                      const __bf16* __restrict__ Bt,
                      const float*  __restrict__ bias,
                      void* __restrict__ Dout,
                      int Mrows) {
    __shared__ __align__(16) __bf16 tA[16 * KP];   // LDS offset 0

    const int row0 = blockIdx.x * 16;

#if USE_TDM
    // ---- Tensor Data Mover: DMA the 16xKP bf16 tile global -> LDS ----
    if ((threadIdx.x >> 5) == 0) {                 // one wave issues the DMA
        const uint64_t ga = (uint64_t)(uintptr_t)(A + (size_t)row0 * KP);
        const uint32_t rows_rem = (uint32_t)(Mrows - row0);

        u32x4 g0;
        g0[0] = 1u;                                   // count=1, user mode
        g0[1] = 0u;                                   // lds_addr = 0 (tA)
        g0[2] = (uint32_t)ga;                         // global_addr[31:0]
        g0[3] = ((uint32_t)(ga >> 32) & 0x01FFFFFFu)  // global_addr[56:32]
                | 0x80000000u;                        // type = 2 ("image")

        i32x8 g1;
        g1[0] = (int)(1u << 16);                      // data_size = 1 (2 bytes)
        g1[1] = (int)(((uint32_t)KP & 0xFFFFu) << 16);          // tensor_dim0 lo
        g1[2] = (int)((((uint32_t)KP >> 16) & 0xFFFFu)          // tensor_dim0 hi
                      | ((rows_rem & 0xFFFFu) << 16));          // tensor_dim1 lo
        g1[3] = (int)(((rows_rem >> 16) & 0xFFFFu)              // tensor_dim1 hi
                      | (((uint32_t)KP & 0xFFFFu) << 16));      // tile_dim0
        g1[4] = (int)16u;                             // tile_dim1=16, tile_dim2=0
        g1[5] = (int)(uint32_t)KP;                    // tensor_dim0_stride lo
        g1[6] = 0;
        g1[7] = 0;

        i32x4 gz4 = {0, 0, 0, 0};
        i32x8 gz8 = {0, 0, 0, 0, 0, 0, 0, 0};
        __builtin_amdgcn_tensor_load_to_lds(g0, g1, gz4, gz4, gz8, 0);
        __builtin_amdgcn_s_wait_tensorcnt(0);
    }
#else
    // ---- fallback: cooperative vector load ----
    constexpr int UNITS = 16 * KP / 2;
    const uint32_t* Asrc = (const uint32_t*)A;
    uint32_t* Adst = (uint32_t*)tA;
    for (int u = threadIdx.x; u < UNITS; u += 256) {
        int r = u / (KP / 2);
        int c = u - r * (KP / 2);
        int grow = row0 + r;
        uint32_t val = 0;
        if (grow < Mrows) val = Asrc[(size_t)grow * (KP / 2) + c];
        Adst[u] = val;
    }
#endif
    __syncthreads();

    const int wave = threadIdx.x >> 5;
    const int lane = threadIdx.x & 31;
    const int m    = lane & 15;
    const int hi   = lane >> 4;
    const int n    = wave * 16 + (lane & 15);   // global output column

    v8f acc;
    const float b = bias ? bias[n] : 0.0f;
#pragma unroll
    for (int r = 0; r < 8; ++r) acc[r] = b;

    constexpr int KSTEPS = KP / 32;
    const uint4* Bp = (const uint4*)(Bt + (size_t)n * KP);

#pragma unroll
    for (int s = 0; s < KSTEPS; ++s) {
        FragAB afr, bfr;
        // A fragment (16x32): hi=0 -> K=32s+[0..7],[16..23]; hi=1 -> +8
        const __bf16* ap = tA + m * KP + 32 * s + 8 * hi;
        afr.u[0] = *(const uint4*)(ap);
        afr.u[1] = *(const uint4*)(ap + 16);
        // B fragment (32x16): col n, hi=0 -> K=32s+[0..7],[16..23]; hi=1 -> +8
        bfr.u[0] = Bp[4 * s + hi];
        bfr.u[1] = Bp[4 * s + 2 + hi];
        acc = __builtin_amdgcn_wmma_f32_16x16x32_bf16(
            false, afr.v, false, bfr.v, (short)0, acc, false, false);
    }

    // D layout: VGPR r, lanes 0-15 -> M=r ; lanes 16-31 -> M=8+r ; N=lane&15
    if (OUT_BF16) {
        __bf16* D = (__bf16*)Dout;
        if (row0 + 16 <= Mrows) {                    // full tile: no guards
#pragma unroll
            for (int r = 0; r < 8; ++r)
                D[(size_t)(row0 + r + 8 * hi) * 128 + n] = f2bf(acc[r]);
        } else {
#pragma unroll
            for (int r = 0; r < 8; ++r) {
                int row = row0 + r + 8 * hi;
                if (row < Mrows) D[(size_t)row * 128 + n] = f2bf(acc[r]);
            }
        }
    } else {
        float* D = (float*)Dout;
        if (row0 + 16 <= Mrows) {
#pragma unroll
            for (int r = 0; r < 8; ++r)
                D[(size_t)(row0 + r + 8 * hi) * 128 + n] = acc[r];
        } else {
#pragma unroll
            for (int r = 0; r < 8; ++r) {
                int row = row0 + r + 8 * hi;
                if (row < Mrows) D[(size_t)row * 128 + n] = acc[r];
            }
        }
    }
}

// ---------------------------------------------------------------------------
// Pass A: per-edge attention scores + segment max (atomic on encoded float)
// ---------------------------------------------------------------------------
__global__ __launch_bounds__(256)
void scores_kernel(const __bf16* __restrict__ qb, const __bf16* __restrict__ kb,
                   const __bf16* __restrict__ eproj,
                   const int* __restrict__ src, const int* __restrict__ dst,
                   float* __restrict__ scores, unsigned* __restrict__ m_enc,
                   int E) {
    int wave = threadIdx.x >> 5;
    int lane = threadIdx.x & 31;
    int e = blockIdx.x * 8 + wave;
    if (e >= E) return;

    int s_n = src[e];
    int d_n = dst[e];
    int c   = lane * 4;

    uint2 qu = *(const uint2*)(qb    + (size_t)d_n * 128 + c);
    uint2 ku = *(const uint2*)(kb    + (size_t)s_n * 128 + c);
    uint2 eu = *(const uint2*)(eproj + (size_t)e   * 128 + c);

    float p = bf_lo(qu.x) * (bf_lo(ku.x) + bf_lo(eu.x))
            + bf_hi(qu.x) * (bf_hi(ku.x) + bf_hi(eu.x))
            + bf_lo(qu.y) * (bf_lo(ku.y) + bf_lo(eu.y))
            + bf_hi(qu.y) * (bf_hi(ku.y) + bf_hi(eu.y));

#pragma unroll
    for (int off = 1; off < 8; off <<= 1)
        p += __shfl_xor(p, off, 32);

    if ((lane & 7) == 0) {
        int head = lane >> 3;
        float sc = p * SCALE_F;
        scores[(size_t)e * HEADS + head] = sc;
        atomicMax(m_enc + (size_t)d_n * HEADS + head, enc_f32(sc));
    }
}

// ---------------------------------------------------------------------------
// Pass B: a = exp(score - max), denom += a
// ---------------------------------------------------------------------------
__global__ void expdenom_kernel(float* __restrict__ scores,
                                const unsigned* __restrict__ m_enc,
                                float* __restrict__ denom,
                                const int* __restrict__ dst, int E) {
    size_t tid = (size_t)blockIdx.x * blockDim.x + threadIdx.x;
    if (tid >= (size_t)E * HEADS) return;
    size_t e = tid >> 2;
    int    h = (int)(tid & 3);
    int d_n = dst[e];
    float mm = dec_f32(m_enc[(size_t)d_n * HEADS + h]);
    float a = __expf(scores[tid] - mm);
    scores[tid] = a;
    atomicAdd(denom + (size_t)d_n * HEADS + h, a);
}

// ---------------------------------------------------------------------------
// Pass C: out[dst] += alpha * (v[src] + e_proj)   (out pre-loaded with skip)
// ---------------------------------------------------------------------------
__global__ __launch_bounds__(256)
void aggregate_kernel(const __bf16* __restrict__ vb, const __bf16* __restrict__ eproj,
                      const float* __restrict__ a, const float* __restrict__ denom,
                      const int* __restrict__ src, const int* __restrict__ dst,
                      float* __restrict__ out, int E) {
    int wave = threadIdx.x >> 5;
    int lane = threadIdx.x & 31;
    int e = blockIdx.x * 8 + wave;
    if (e >= E) return;

    int s_n = src[e];
    int d_n = dst[e];
    int c    = lane * 4;
    int head = lane >> 3;

    float alpha = a[(size_t)e * HEADS + head] /
                  denom[(size_t)d_n * HEADS + head];

    uint2 vu = *(const uint2*)(vb    + (size_t)s_n * 128 + c);
    uint2 eu = *(const uint2*)(eproj + (size_t)e   * 128 + c);

    float* o = out + (size_t)d_n * 128 + c;
    atomicAdd(o + 0, alpha * (bf_lo(vu.x) + bf_lo(eu.x)));
    atomicAdd(o + 1, alpha * (bf_hi(vu.x) + bf_hi(eu.x)));
    atomicAdd(o + 2, alpha * (bf_lo(vu.y) + bf_lo(eu.y)));
    atomicAdd(o + 3, alpha * (bf_hi(vu.y) + bf_hi(eu.y)));
}

// ---------------------------------------------------------------------------
// Host-side orchestration
// ---------------------------------------------------------------------------
extern "C" void kernel_launch(void* const* d_in, const int* in_sizes, int n_in,
                              void* d_out, int out_size, void* d_ws, size_t ws_size,
                              hipStream_t stream) {
    const float* memory_table = (const float*)d_in[0];
    const float* edge_attr    = (const float*)d_in[1];
    const int*   n_id         = (const int*)d_in[2];
    const int*   edge_index   = (const int*)d_in[3];

    const int N = in_sizes[2];
    const int E = in_sizes[3] / 2;

    const int* srcI = edge_index;
    const int* dstI = edge_index + E;

    const float* P[2][9];
    for (int l = 0; l < 2; ++l)
        for (int j = 0; j < 9; ++j)
            P[l][j] = (const float*)d_in[4 + l * 9 + j];

    // ----- workspace carve-out -----
    size_t off = 0;
    auto carve = [&](size_t bytes) -> void* {
        void* p = (char*)d_ws + off;
        off += (bytes + 255) & ~(size_t)255;
        return p;
    };
    __bf16* xb    = (__bf16*)carve((size_t)N * 128 * 2);
    __bf16* qb    = (__bf16*)carve((size_t)N * 128 * 2);
    __bf16* kb    = (__bf16*)carve((size_t)N * 128 * 2);
    __bf16* vb    = (__bf16*)carve((size_t)N * 128 * 2);
    __bf16* eb    = (__bf16*)carve((size_t)E * EDGE_KP * 2);
    __bf16* eproj = (__bf16*)carve((size_t)E * 128 * 2);
    float*  hacc  = (float*) carve((size_t)N * 128 * 4);
    float*  sc    = (float*) carve((size_t)E * HEADS * 4);
    unsigned* mEn = (unsigned*)carve((size_t)N * HEADS * 4);
    float*  den   = (float*) carve((size_t)N * HEADS * 4);
    __bf16* wq_t  = (__bf16*)carve(128 * 128 * 2);
    __bf16* wk_t  = (__bf16*)carve(128 * 128 * 2);
    __bf16* wv_t  = (__bf16*)carve(128 * 128 * 2);
    __bf16* ws_t  = (__bf16*)carve(128 * 128 * 2);
    __bf16* we_t  = (__bf16*)carve(128 * EDGE_KP * 2);

    const int TB = 256;
    const int mtilesN = (N + 15) / 16;
    const int mtilesE = (E + 15) / 16;
    const int edgeBlks = (E + 7) / 8;

    gathercvt_kernel<<<(N * 32 + TB - 1) / TB, TB, 0, stream>>>(memory_table, n_id, xb, N);
    {
        size_t tot = (size_t)E * EDGE_KP;
        padcvt_kernel<<<(unsigned)((tot + TB - 1) / TB), TB, 0, stream>>>(
            edge_attr, eb, E, EDGE_D, EDGE_KP);
    }

    const __bf16* layer_in = xb;
    for (int l = 0; l < 2; ++l) {
        const float *Wq = P[l][0], *Bq = P[l][1], *Wk = P[l][2], *Bk = P[l][3],
                    *Wv = P[l][4], *Bv = P[l][5], *We = P[l][6], *Ws = P[l][7],
                    *Bs = P[l][8];
        float* outAcc = (l == 0) ? hacc : (float*)d_out;

        int wn = (128 * 128 + TB - 1) / TB;
        wconv_kernel<<<wn, TB, 0, stream>>>(Wq, wq_t, 128, 128);
        wconv_kernel<<<wn, TB, 0, stream>>>(Wk, wk_t, 128, 128);
        wconv_kernel<<<wn, TB, 0, stream>>>(Wv, wv_t, 128, 128);
        wconv_kernel<<<wn, TB, 0, stream>>>(Ws, ws_t, 128, 128);
        wconv_kernel<<<(128 * EDGE_KP + TB - 1) / TB, TB, 0, stream>>>(
            We, we_t, EDGE_D, EDGE_KP);

        gemm_bf16_kernel<128, 1><<<mtilesN, TB, 0, stream>>>(layer_in, wq_t, Bq, qb, N);
        gemm_bf16_kernel<128, 1><<<mtilesN, TB, 0, stream>>>(layer_in, wk_t, Bk, kb, N);
        gemm_bf16_kernel<128, 1><<<mtilesN, TB, 0, stream>>>(layer_in, wv_t, Bv, vb, N);
        gemm_bf16_kernel<128, 0><<<mtilesN, TB, 0, stream>>>(layer_in, ws_t, Bs, outAcc, N);
        gemm_bf16_kernel<EDGE_KP, 1><<<mtilesE, TB, 0, stream>>>(eb, we_t, nullptr, eproj, E);

        (void)hipMemsetAsync(mEn, 0, (size_t)N * HEADS * 4, stream);
        (void)hipMemsetAsync(den, 0, (size_t)N * HEADS * 4, stream);
        scores_kernel<<<edgeBlks, TB, 0, stream>>>(qb, kb, eproj, srcI, dstI, sc, mEn, E);
        expdenom_kernel<<<(unsigned)(((size_t)E * HEADS + TB - 1) / TB), TB, 0, stream>>>(
            sc, mEn, den, dstI, E);
        aggregate_kernel<<<edgeBlks, TB, 0, stream>>>(vb, eproj, sc, den, srcI, dstI,
                                                      outAcc, E);

        if (l == 0) {
            size_t tot = (size_t)N * 128;
            relucvt_kernel<<<(unsigned)((tot + TB - 1) / TB), TB, 0, stream>>>(hacc, xb, tot);
            layer_in = xb;
        }
    }
}
